// XCA_label_8916352106921
// MI455X (gfx1250) — matrix operand order repaired and checked by reading, not compiled, not run on hardware
//
#include <hip/hip_runtime.h>
#include <hip/hip_bf16.h>

typedef __attribute__((ext_vector_type(16))) _Float16 v16h;
typedef __attribute__((ext_vector_type(8)))  float    v8f;

#define BB   8
#define NN   4096
#define CC   768
#define CC2  1536
#define HH   6
#define HDIM 128
#define FEPS 1e-12f

// LDS row padding: row stride (halves) == 8 mod 128 -> 16B fragment reads from
// 16 consecutive lanes hit 64 distinct banks.
#define WPAD 776    // 768 + 8
#define QPAD 1032   // 1024 + 8

// ---------------- CDNA5 async copy: global -> LDS (ASYNCcnt-tracked) --------
__device__ __forceinline__ void async_cp16(const _Float16* lds_dst,
                                           const _Float16* gsrc) {
    uint32_t loff = (uint32_t)(uintptr_t)lds_dst;  // low 32 bits = LDS offset
    asm volatile("global_load_async_to_lds_b128 %0, %1, off"
                 :: "v"(loff), "v"(gsrc)
                 : "memory");
}
__device__ __forceinline__ void async_wait0() {
    asm volatile("s_wait_asynccnt 0x0" ::: "memory");
}

// ---------------- WMMA fragment loaders (16x16x32 f16; contiguous k order, valid as A and B match) ----------
// A (16x32, M x K): lane holds row M=lane%16; halves 0..15 -> K=16*grp+0..15
//                   (grp = lane/16)
__device__ __forceinline__ v16h ldA(const _Float16* p, int grp) {
    const _Float16* q = p + grp * 16;
    v16h r;
#pragma unroll
    for (int i = 0; i < 16; ++i) r[i] = q[i];
    return r;
}
// B (32x16, K x N): lane holds column N=lane%16; halves 0..15 -> K=16*grp+0..15
__device__ __forceinline__ v16h ldB(const _Float16* p, int grp) {
    const _Float16* q = p + grp * 16;
    v16h r;
#pragma unroll
    for (int i = 0; i < 16; ++i) r[i] = q[i];
    return r;
}
__device__ __forceinline__ v8f wmma_f16(v16h a, v16h b, v8f c) {
    return __builtin_amdgcn_wmma_f32_16x16x32_f16(false, a, false, b, (short)0, c,
                                                  false, false);
}

// ---------------- prep kernels ----------------------------------------------
// W: [K][Nc] row-major f32  ->  Wt: [Nc][K] f16
__global__ void k_transpose_w(const float* __restrict__ W, _Float16* __restrict__ Wt,
                              int K, int Nc) {
    int idx = blockIdx.x * blockDim.x + threadIdx.x;
    if (idx < K * Nc) {
        int j = idx / K;
        int k = idx % K;
        Wt[idx] = (_Float16)W[(size_t)k * Nc + j];
    }
}
__global__ void k_f32_to_f16(const float* __restrict__ src, _Float16* __restrict__ dst,
                             int n) {
    int idx = blockIdx.x * blockDim.x + threadIdx.x;
    if (idx < n) dst[idx] = (_Float16)src[idx];
}

// ---------------- stage 1: qkv = x @ W_qkv, write q/k [B,H,hd,N] f16 + sumsq --
// Block-shared 16x768 weight slab async-staged into LDS once; 4 waves share it.
// B-stream reads pre-converted f16 x (xh) -> no in-loop cvt, half the bytes.
__global__ void k_qkv(const _Float16* __restrict__ xh, const _Float16* __restrict__ WqkvT,
                      _Float16* __restrict__ qh, _Float16* __restrict__ kh,
                      float* __restrict__ sumsq) {
    __shared__ _Float16 wsh[16 * WPAD];  // ~24.8 KB
    const int tidx = threadIdx.x;
    const int lane = tidx & 31, w = tidx >> 5;
    const int lane16 = lane & 15, grp = lane >> 4;
    const int nt = blockIdx.x * 4 + w;   // token tile
    const int jt = blockIdx.y;           // channel tile (0..95)
    const int b  = blockIdx.z;
    const int n  = nt * 16 + lane16;

    // async stage: 16 rows x 768 halves = 1536 x 16B chunks, 128 threads
#pragma unroll
    for (int i = 0; i < 12; ++i) {
        int ch  = tidx + 128 * i;
        int row = ch / 96;          // 96 chunks per row
        int col = (ch % 96) * 8;
        async_cp16(&wsh[row * WPAD + col],
                   WqkvT + (size_t)(jt * 16 + row) * CC + col);
    }
    async_wait0();
    __syncthreads();

    const _Float16* arow = wsh + lane16 * WPAD;              // LDS A-slab
    const _Float16* brow = xh + ((size_t)b * NN + n) * CC;   // global B-stream (f16)
    v8f acc = {};
#pragma unroll 4
    for (int k0 = 0; k0 < CC; k0 += 32)
        acc = wmma_f16(ldA(arow + k0, grp), ldB(brow + k0, grp), acc);

#pragma unroll
    for (int r = 0; r < 8; ++r) {
        int j = jt * 16 + r + 8 * grp;
        float val = acc[r];
        int isk = (j >= CC) ? 1 : 0;
        int c = isk ? j - CC : j;
        int h = c / HDIM, hc = c % HDIM;
        _Float16* dst = (isk ? kh : qh) +
                        (((size_t)b * HH + h) * HDIM + hc) * (size_t)NN + n;
        *dst = (_Float16)val;
        float ss = val * val;  // reduce over the 16 tokens of this half-wave
        ss += __shfl_xor(ss, 1, 32);
        ss += __shfl_xor(ss, 2, 32);
        ss += __shfl_xor(ss, 4, 32);
        ss += __shfl_xor(ss, 8, 32);
        if (lane16 == 0)
            atomicAdd(&sumsq[((size_t)isk * BB + b) * CC + c], ss);
    }
}

// ---------------- stage 2: attn = (q/|q|)(k/|k|)^T * temp, [B,H,128,128] f32 -
// cti is block-uniform: the 16 q-rows are async-staged into LDS in 4 chunks of
// 16x1024 f16 and shared by all 8 waves (each wave owns one ctj).
__global__ void k_attn(const _Float16* __restrict__ qh, const _Float16* __restrict__ kh,
                       const float* __restrict__ sumsq, const float* __restrict__ temp,
                       float* __restrict__ attn) {
    __shared__ _Float16 qs[16 * QPAD];  // ~33 KB
    const int tidx = threadIdx.x;
    const int lane = tidx & 31, w = tidx >> 5;
    const int lane16 = lane & 15, grp = lane >> 4;
    const int bh = blockIdx.z, b = bh / HH, h = bh % HH;
    const int cti = blockIdx.x;  // 0..7 (shared by block)
    const int ctj = w;           // 0..7 (one per wave)

    const _Float16* qbase = qh + (size_t)bh * HDIM * NN + (size_t)cti * 16 * NN;
    const _Float16* brow  = kh + (size_t)bh * HDIM * NN +
                            (size_t)(ctj * 16 + lane16) * NN;
    v8f acc = {};
    for (int c = 0; c < 4; ++c) {
        // stage 16 rows x 1024 halves = 2048 x 16B chunks, 256 threads
#pragma unroll
        for (int i = 0; i < 8; ++i) {
            int ch  = tidx + 256 * i;
            int row = ch >> 7;            // 128 chunks per row
            int col = (ch & 127) * 8;
            async_cp16(&qs[row * QPAD + col],
                       qbase + (size_t)row * NN + c * 1024 + col);
        }
        async_wait0();
        __syncthreads();

        const _Float16* arow = qs + lane16 * QPAD;
        const _Float16* bk   = brow + c * 1024;
        if (c < 3) __builtin_prefetch(bk + 1024, 0, 0);  // next k-chunk
#pragma unroll 4
        for (int k0 = 0; k0 < 1024; k0 += 32)
            acc = wmma_f16(ldA(arow + k0, grp), ldB(bk + k0, grp), acc);
        __syncthreads();  // before next chunk overwrites qs
    }

    const int cj = ctj * 16 + lane16;
    float nk   = sqrtf(sumsq[((size_t)BB + b) * CC + h * HDIM + cj]);
    float invk = 1.0f / fmaxf(nk, FEPS);
    float tm   = temp[h];
#pragma unroll
    for (int r = 0; r < 8; ++r) {
        int ci = cti * 16 + r + 8 * grp;
        float nq   = sqrtf(sumsq[(size_t)b * CC + h * HDIM + ci]);
        float invq = 1.0f / fmaxf(nq, FEPS);
        attn[((size_t)bh * HDIM + ci) * HDIM + cj] = acc[r] * invq * invk * tm;
    }
}

// ---------------- stage 3: row softmax over 128, emit f16 --------------------
__global__ void k_softmax(const float* __restrict__ attn, _Float16* __restrict__ attnh) {
    const int row = blockIdx.x;            // B*H*128 rows
    const int tid = threadIdx.x;           // 128 threads
    const int lane = tid & 31, wid = tid >> 5;
    __shared__ float red[4];
    float v = attn[(size_t)row * HDIM + tid];
    float m = v;
    m = fmaxf(m, __shfl_xor(m, 16, 32));
    m = fmaxf(m, __shfl_xor(m, 8, 32));
    m = fmaxf(m, __shfl_xor(m, 4, 32));
    m = fmaxf(m, __shfl_xor(m, 2, 32));
    m = fmaxf(m, __shfl_xor(m, 1, 32));
    if (lane == 0) red[wid] = m;
    __syncthreads();
    m = fmaxf(fmaxf(red[0], red[1]), fmaxf(red[2], red[3]));
    float e = __expf(v - m);
    float s = e;
    s += __shfl_xor(s, 16, 32);
    s += __shfl_xor(s, 8, 32);
    s += __shfl_xor(s, 4, 32);
    s += __shfl_xor(s, 2, 32);
    s += __shfl_xor(s, 1, 32);
    __syncthreads();
    if (lane == 0) red[wid] = s;
    __syncthreads();
    s = red[0] + red[1] + red[2] + red[3];
    attnh[(size_t)row * HDIM + tid] = (_Float16)(e / s);
}

// ---------------- stage 4: out = attn @ v -> [B,N,C] f16 (coalesced) ---------
// A rows = tokens from yh in native [B,N,C] layout (fragment layout absorbs the
// v-transpose), B cols = output channels from attn_h.
__global__ void k_av(const _Float16* __restrict__ attnh, const _Float16* __restrict__ yh,
                     _Float16* __restrict__ outh) {
    const int lane = threadIdx.x & 31, w = threadIdx.x >> 5;
    const int lane16 = lane & 15, grp = lane >> 4;
    const int bh = blockIdx.z, b = bh / HH, h = bh % HH;
    const int nt  = blockIdx.x * 8 + w;    // 0..255
    const int cit = blockIdx.y;            // 0..7
    const _Float16* arow = yh + ((size_t)b * NN + nt * 16 + lane16) * CC + h * HDIM;
    const _Float16* brow = attnh + ((size_t)bh * HDIM + cit * 16 + lane16) * HDIM;
    v8f acc = {};
#pragma unroll
    for (int k0 = 0; k0 < HDIM; k0 += 32)
        acc = wmma_f16(ldA(arow + k0, grp), ldB(brow + k0, grp), acc);
#pragma unroll
    for (int r = 0; r < 8; ++r) {
        int n  = nt * 16 + r + 8 * grp;
        int ci = cit * 16 + lane16;
        outh[((size_t)b * NN + n) * CC + h * HDIM + ci] = (_Float16)acc[r];
    }
}

// ---------------- stage 5: final = out @ W_proj + b_proj ---------------------
// Block-shared 16x768 W_projT slab (the B-side columns) async-staged into LDS.
__global__ void k_proj(const _Float16* __restrict__ outh, const _Float16* __restrict__ WprojT,
                       const float* __restrict__ bproj, float* __restrict__ out) {
    __shared__ _Float16 wsh[16 * WPAD];  // ~24.8 KB
    const int tidx = threadIdx.x;
    const int lane = tidx & 31, w = tidx >> 5;
    const int lane16 = lane & 15, grp = lane >> 4;
    const int mt = blockIdx.x * 8 + w;     // token tile 0..2047
    const int jt = blockIdx.y;             // 0..47

#pragma unroll
    for (int i = 0; i < 6; ++i) {          // 1536 x 16B chunks, 256 threads
        int ch  = tidx + 256 * i;
        int row = ch / 96;
        int col = (ch % 96) * 8;
        async_cp16(&wsh[row * WPAD + col],
                   WprojT + (size_t)(jt * 16 + row) * CC + col);
    }
    async_wait0();
    __syncthreads();

    const _Float16* arow = outh + (size_t)(mt * 16 + lane16) * CC;
    const _Float16* brow = wsh + lane16 * WPAD;
    v8f acc = {};
#pragma unroll 4
    for (int k0 = 0; k0 < CC; k0 += 32)
        acc = wmma_f16(ldA(arow + k0, grp), ldB(brow + k0, grp), acc);

    const int j = jt * 16 + lane16;
    const float bias = bproj[j];
#pragma unroll
    for (int r = 0; r < 8; ++r) {
        int tok = mt * 16 + r + 8 * grp;
        out[(size_t)tok * CC + j] = acc[r] + bias;   // coalesced f32 stores
    }
}

extern "C" void kernel_launch(void* const* d_in, const int* in_sizes, int n_in,
                              void* d_out, int out_size, void* d_ws, size_t ws_size,
                              hipStream_t stream) {
    const float* x     = (const float*)d_in[0];
    const float* y     = (const float*)d_in[1];
    const float* Wqkv  = (const float*)d_in[2];
    const float* temp  = (const float*)d_in[3];
    const float* Wproj = (const float*)d_in[4];
    const float* bproj = (const float*)d_in[5];
    float* out = (float*)d_out;

    char* ws = (char*)d_ws;
    size_t off = 0;
    auto alloc = [&](size_t bytes) -> void* {
        void* p = (void*)(ws + off);
        off += (bytes + 255) & ~(size_t)255;
        return p;
    };
    _Float16* WqkvT  = (_Float16*)alloc((size_t)CC2 * CC * 2);              // 2.25 MB
    _Float16* WprojT = (_Float16*)alloc((size_t)CC * CC * 2);               // 1.1 MB
    _Float16* xh     = (_Float16*)alloc((size_t)BB * NN * CC * 2);          // 50 MB
    _Float16* qh     = (_Float16*)alloc((size_t)BB * HH * HDIM * NN * 2);   // 50 MB
    _Float16* kh     = (_Float16*)alloc((size_t)BB * HH * HDIM * NN * 2);   // 50 MB
    _Float16* yh     = (_Float16*)alloc((size_t)BB * NN * CC * 2);          // 50 MB
    float*    sumsq  = (float*)   alloc((size_t)2 * BB * CC * 4);           // 48 KB
    float*    attn   = (float*)   alloc((size_t)BB * HH * HDIM * HDIM * 4); // 3 MB
    _Float16* attnh  = (_Float16*)alloc((size_t)BB * HH * HDIM * HDIM * 2); // 1.5 MB
    _Float16* outh   = (_Float16*)alloc((size_t)BB * NN * CC * 2);          // 50 MB

    hipMemsetAsync(sumsq, 0, (size_t)2 * BB * CC * 4, stream);

    k_transpose_w<<<(CC * CC2 + 255) / 256, 256, 0, stream>>>(Wqkv, WqkvT, CC, CC2);
    k_transpose_w<<<(CC * CC + 255) / 256, 256, 0, stream>>>(Wproj, WprojT, CC, CC);
    {
        int n = BB * NN * CC;
        k_f32_to_f16<<<(n + 255) / 256, 256, 0, stream>>>(x, xh, n);
        k_f32_to_f16<<<(n + 255) / 256, 256, 0, stream>>>(y, yh, n);
    }
    // qkv: grid (token-tiles/4, channel-tiles, B), 4 waves per block
    k_qkv<<<dim3(NN / 64, CC2 / 16, BB), 128, 0, stream>>>(xh, WqkvT, qh, kh, sumsq);
    // attention scores: cti per block, ctj per wave (8 waves)
    k_attn<<<dim3(8, 1, BB * HH), 256, 0, stream>>>(qh, kh, sumsq, temp, attn);
    k_softmax<<<BB * HH * HDIM, 128, 0, stream>>>(attn, attnh);
    k_av<<<dim3(NN / 128, HDIM / 16, BB * HH), 256, 0, stream>>>(attnh, yh, outh);
    k_proj<<<dim3(BB * NN / 128, CC / 16, 1), 256, 0, stream>>>(outh, WprojT, bproj, out);
}
